// TransformerBlock_67293547594159
// MI455X (gfx1250) — matrix operand order repaired and checked by reading, not compiled, not run on hardware
//
#include <hip/hip_runtime.h>
#include <cstdint>
#include <cstddef>

// ---------------------------------------------------------------------------
// Types for CDNA5 WMMA (gfx1250, wave32)
// ---------------------------------------------------------------------------
typedef __attribute__((ext_vector_type(16))) __bf16       v16bf;
typedef __attribute__((ext_vector_type(8)))  float        v8f;
typedef __attribute__((ext_vector_type(4)))  int          v4i_t;
typedef __attribute__((ext_vector_type(8)))  int          v8i_t;
typedef __attribute__((ext_vector_type(4)))  unsigned int v4u_t;

struct __align__(16) U4 { unsigned int x, y, z, w; };

union Frag16 {
    v16bf bf;
    U4    u[2];
};

typedef __attribute__((address_space(1))) v4i_t          as1_v4i;
typedef __attribute__((address_space(3))) v4i_t          as3_v4i;
typedef __attribute__((address_space(3))) unsigned short as3_u16;

__device__ __forceinline__ float bf2f(unsigned short u) {
    unsigned int x = ((unsigned int)u) << 16;
    float f;
    __builtin_memcpy(&f, &x, 4);
    return f;
}
__device__ __forceinline__ unsigned short f2bf(float f) {
    unsigned int x;
    __builtin_memcpy(&x, &f, 4);
    unsigned int r = x + 0x7FFFu + ((x >> 16) & 1u);   // round-to-nearest-even
    return (unsigned short)(r >> 16);
}

__device__ __forceinline__ v8f wmma_bf16(v16bf a, v16bf b, v8f c) {
    // (neg_a, A, neg_b, B, c_mod, C, reuse_a, reuse_b)
    return __builtin_amdgcn_wmma_f32_16x16x32_bf16(false, a, false, b, (short)0, c,
                                                   false, false);
}

// ---------------------------------------------------------------------------
// CDNA5 async global->LDS copy (16B per lane), tracked by ASYNCcnt.
// ---------------------------------------------------------------------------
__device__ __forceinline__ void async_copy16(const unsigned short* g, unsigned short* l) {
#if __has_builtin(__builtin_amdgcn_global_load_async_to_lds_b128)
    __builtin_amdgcn_global_load_async_to_lds_b128((as1_v4i*)g, (as3_v4i*)l, 0, 0);
#else
    const unsigned lds_off = (unsigned)(size_t)(as3_u16*)l;
    asm volatile("global_load_async_to_lds_b128 %0, %1, off"
                 :: "v"(lds_off), "v"((unsigned long long)(size_t)g)
                 : "memory");
#endif
}

template <int N>
__device__ __forceinline__ void wait_async() {
#if __has_builtin(__builtin_amdgcn_s_wait_asynccnt)
    __builtin_amdgcn_s_wait_asynccnt(N);
#else
    asm volatile("s_wait_asynccnt %0" :: "i"(N) : "memory");
#endif
}

template <int N>
__device__ __forceinline__ void wait_tensor() {
#if __has_builtin(__builtin_amdgcn_s_wait_tensorcnt)
    __builtin_amdgcn_s_wait_tensorcnt((unsigned short)N);
#else
    asm volatile("s_wait_tensorcnt %0" :: "i"(N) : "memory");
#endif
}

// ---------------------------------------------------------------------------
// Tensor Data Mover: 2D bf16 tile (tile_w x tile_h) from row-major tensor
// (row length stride_elems) into LDS at lds_addr, with LDS padding of 4 dwords
// after every 32 dwords stored (row pitch 64 -> 72 elements, tile_w must be 64).
// D# layout per CDNA5 ISA 8.3-8.6.  Toolchain uses the 6-arg builtin:
// (uint32x4 g0, int32x8 g1, int32x4 g2, int32x4 g3, int32x8 g4, i32 cpol)
// ---------------------------------------------------------------------------
#define HAVE_TDM __has_builtin(__builtin_amdgcn_tensor_load_to_lds)

#if HAVE_TDM
__device__ __forceinline__ void tdm_load_2d(const unsigned short* gtile,
                                            unsigned lds_addr,
                                            unsigned tensor_w, unsigned tensor_h,
                                            unsigned stride_elems,
                                            unsigned tile_w, unsigned tile_h) {
    const unsigned long long ga = (unsigned long long)(size_t)gtile;
    v4u_t g0;
    g0[0] = 1u;                                              // count=1, user D#
    g0[1] = lds_addr;                                        // [63:32] lds_addr
    g0[2] = (unsigned)ga;                                    // global_addr low
    g0[3] = (unsigned)((ga >> 32) & 0x01FFFFFFu) | (2u << 30);  // addr hi | type=2
    v8i_t g1;
    g1[0] = (int)((1u << 16)      // data_size = 2 bytes
                  | (1u << 20)    // pad_enable
                  | (4u << 22)    // pad_interval: every 32 dwords (128B)
                  | (3u << 25));  // pad_amount: 4 dwords (16B)
    g1[1] = (int)((tensor_w & 0xFFFFu) << 16);                       // dim0 [47:32]=0, [63:48]
    g1[2] = (int)(((tensor_w >> 16) & 0xFFFFu) | ((tensor_h & 0xFFFFu) << 16));
    g1[3] = (int)(((tensor_h >> 16) & 0xFFFFu) | ((tile_w & 0xFFFFu) << 16));
    g1[4] = (int)(tile_h & 0xFFFFu);                                 // tile_dim1, tile_dim2=0
    g1[5] = (int)stride_elems;                                       // dim0 stride low
    g1[6] = 0;                                                       // stride hi, dim1 stride
    g1[7] = 0;
    const v4i_t gz4 = {0, 0, 0, 0};
    const v8i_t gz8 = {0, 0, 0, 0, 0, 0, 0, 0};
    __builtin_amdgcn_tensor_load_to_lds(g0, g1, gz4, gz4, gz8, 0);
}
#endif

// A fragment: 16(M) x 32(K).  Lanes 0-15: row = lane, K in {0..7, 16..23};
// lanes 16-31: row = lane-16, K in {8..15, 24..31}.
__device__ __forceinline__ v16bf load_frag_a(const unsigned short* p, int row_stride,
                                             int lane) {
    const int r     = lane & 15;
    const int khalf = (lane >> 4) & 1;
    const unsigned short* q = p + (size_t)r * row_stride + khalf * 8;
    Frag16 f;
    f.u[0] = *(const U4*)(q);        // K + {0..7}  (or {8..15})
    f.u[1] = *(const U4*)(q + 16);   // K + {16..23} (or {24..31})
    return f.bf;
}

// B fragment: 32(K) x 16(N), operand stored as rows of length K (weight [N x K]).
// Lanes 0-15: col = lane, K in {0..7, 16..23}; lanes 16-31: col = lane-16, K in {8..15, 24..31}.
__device__ __forceinline__ v16bf load_frag_b(const unsigned short* p, int row_stride,
                                             int lane) {
    const int n     = lane & 15;
    const int khalf = (lane >> 4) & 1;
    const unsigned short* q = p + (size_t)n * row_stride + khalf * 8;
    Frag16 f;
    f.u[0] = *(const U4*)(q);
    f.u[1] = *(const U4*)(q + 16);
    return f.bf;
}

// ---------------------------------------------------------------------------
// Problem constants
// ---------------------------------------------------------------------------
#define BB   2
#define SS   2048
#define DIMD 2048
#define HH   16
#define DHH  128
#define FFF  8192
#define MM   (BB * SS)          // 4096 rows

// ---------------------------------------------------------------------------
// fp32 -> bf16 cast (weights)
// ---------------------------------------------------------------------------
__global__ __launch_bounds__(256) void cast_bf16_kernel(const float* __restrict__ in,
                                                        unsigned short* __restrict__ out,
                                                        int n) {
    int i = (blockIdx.x * 256 + threadIdx.x) * 4;
    if (i + 3 < n) {
        out[i + 0] = f2bf(in[i + 0]);
        out[i + 1] = f2bf(in[i + 1]);
        out[i + 2] = f2bf(in[i + 2]);
        out[i + 3] = f2bf(in[i + 3]);
    }
}

// ---------------------------------------------------------------------------
// RMSNorm: one block per row of 2048, output bf16 h = x * rsqrt(mean(x^2)+eps) * g
// ---------------------------------------------------------------------------
__global__ __launch_bounds__(256) void rmsnorm_kernel(const float* __restrict__ x,
                                                      const float* __restrict__ g,
                                                      unsigned short* __restrict__ out) {
    const int row = blockIdx.x, tid = threadIdx.x;
    const float* xr = x + (size_t)row * DIMD;
    float vals[8];
    float ss = 0.f;
#pragma unroll
    for (int i = 0; i < 8; i++) {
        float v = xr[tid + i * 256];
        vals[i] = v;
        ss += v * v;
    }
#pragma unroll
    for (int off = 1; off < 32; off <<= 1) ss += __shfl_xor(ss, off, 32);
    __shared__ float red[8];
    if ((tid & 31) == 0) red[tid >> 5] = ss;
    __syncthreads();
    float tot = 0.f;
#pragma unroll
    for (int i = 0; i < 8; i++) tot += red[i];
    const float inv = rsqrtf(tot * (1.0f / DIMD) + 1e-6f);
    unsigned short* orow = out + (size_t)row * DIMD;
#pragma unroll
    for (int i = 0; i < 8; i++) {
        int c = tid + i * 256;
        orow[c] = f2bf(vals[i] * inv * g[c]);
    }
}

// ---------------------------------------------------------------------------
// RoPE, in place on bf16 tensor laid out [M, H*DH].  One thread per (m,h,d<64).
// ---------------------------------------------------------------------------
__global__ __launch_bounds__(256) void rope_kernel(unsigned short* __restrict__ t,
                                                   int total) {
    int idx = blockIdx.x * 256 + threadIdx.x;
    if (idx >= total) return;
    const int d   = idx & 63;
    const int tmp = idx >> 6;
    const int h   = tmp & (HH - 1);
    const int m   = tmp >> 4;
    const int s   = m & (SS - 1);
    const float freq = __powf(10000.f, -(float)(2 * d) * (1.0f / DHH));
    const float ang  = (float)s * freq;
    const float c = __cosf(ang), sn = __sinf(ang);
    const size_t base = (size_t)m * (HH * DHH) + h * DHH;
    const float x1 = bf2f(t[base + d]);
    const float x2 = bf2f(t[base + d + 64]);
    t[base + d]      = f2bf(x1 * c - x2 * sn);
    t[base + d + 64] = f2bf(x2 * c + x1 * sn);
}

// ---------------------------------------------------------------------------
// Tiled bf16 WMMA GEMM:  C[M,N] = A[M,K] * W[N,K]^T, f32 accumulate.
// Block tile 256(M) x 128(N), BK=64, double-buffered TDM (or async) staging.
// 8 waves arranged 4(M) x 2(N): wave tile 64x64 = 4x4 accumulators,
// 16 wmmas per 8 fragment loads per K-half.
// EPI 0: store bf16          EPI 1: store f32 (acc + resid)
// EPI 2: store bf16 silu(acc) * gate   (gate read & output may alias)
// ---------------------------------------------------------------------------
template <int EPI>
__global__ __launch_bounds__(256) void gemm_bf16_kernel(
    const unsigned short* __restrict__ A, const unsigned short* __restrict__ W,
    const float* __restrict__ resid, const unsigned short* __restrict__ gate,
    void* __restrict__ outv, int M, int N, int K) {
    __shared__ unsigned short sA[2][256 * 72];
    __shared__ unsigned short sW[2][128 * 72];
    const int tid = threadIdx.x, lane = tid & 31, wave = tid >> 5;
    const int wm = wave >> 1;          // 0..3  -> M offset wm*64
    const int wn = wave & 1;           // 0..1  -> N offset wn*64
    const int mBase = blockIdx.y * 256, nBase = blockIdx.x * 128;

    const v8f zero = {};
    v8f acc[4][4];
#pragma unroll
    for (int a = 0; a < 4; a++)
#pragma unroll
        for (int b = 0; b < 4; b++) acc[a][b] = zero;

    auto issue_tile = [&](int k0, int buf) {
#if HAVE_TDM
        if (wave == 0) {
            tdm_load_2d(A + (size_t)mBase * K + k0,
                        (unsigned)(size_t)(as3_u16*)&sA[buf][0],
                        (unsigned)K, (unsigned)M, (unsigned)K, 64u, 256u);
            tdm_load_2d(W + (size_t)nBase * K + k0,
                        (unsigned)(size_t)(as3_u16*)&sW[buf][0],
                        (unsigned)K, (unsigned)N, (unsigned)K, 64u, 128u);
        }
#else
#pragma unroll
        for (int i = 0; i < 8; i++) {
            const int u  = tid + i * 256;
            const int rr = u >> 3;
            const int cc = (u & 7) * 8;
            async_copy16(A + (size_t)(mBase + rr) * K + k0 + cc, &sA[buf][rr * 72 + cc]);
        }
#pragma unroll
        for (int i = 0; i < 4; i++) {
            const int u  = tid + i * 256;
            const int rr = u >> 3;
            const int cc = (u & 7) * 8;
            async_copy16(W + (size_t)(nBase + rr) * K + k0 + cc, &sW[buf][rr * 72 + cc]);
        }
#endif
    };
    auto wait_tiles_keep = [&]() {     // one tile kept in flight
#if HAVE_TDM
        if (wave == 0) wait_tensor<2>();
#else
        wait_async<12>();
#endif
    };
    auto wait_tiles_all = [&]() {
#if HAVE_TDM
        if (wave == 0) wait_tensor<0>();
#else
        wait_async<0>();
#endif
    };

    issue_tile(0, 0);
    int buf = 0;
    for (int k0 = 0; k0 < K; k0 += 64) {
        if (k0 + 64 < K) {
            issue_tile(k0 + 64, buf ^ 1);
            wait_tiles_keep();
        } else {
            wait_tiles_all();
        }
        __syncthreads();                // all waves see this tile in LDS

        const unsigned short* pA = sA[buf];
        const unsigned short* pW = sW[buf];
#pragma unroll
        for (int kk = 0; kk < 64; kk += 32) {
            v16bf af[4];
#pragma unroll
            for (int mi = 0; mi < 4; mi++)
                af[mi] = load_frag_a(pA + (wm * 64 + mi * 16) * 72 + kk, 72, lane);
            v16bf bcur = load_frag_b(pW + (wn * 64) * 72 + kk, 72, lane);
#pragma unroll
            for (int nt = 0; nt < 4; nt++) {
                v16bf bnxt = bcur;
                if (nt < 3)
                    bnxt = load_frag_b(pW + (wn * 64 + (nt + 1) * 16) * 72 + kk, 72, lane);
#pragma unroll
                for (int mi = 0; mi < 4; mi++)
                    acc[mi][nt] = wmma_bf16(af[mi], bcur, acc[mi][nt]);
                bcur = bnxt;
            }
        }
        __syncthreads();                // everyone done reading buf before overwrite
        buf ^= 1;
    }

    const int lhalf = (lane < 16) ? 0 : 8;
    const int col0  = lane & 15;
#pragma unroll
    for (int mi = 0; mi < 4; mi++) {
#pragma unroll
        for (int nt = 0; nt < 4; nt++) {
#pragma unroll
            for (int j = 0; j < 8; j++) {
                const int m = mBase + wm * 64 + mi * 16 + lhalf + j;
                const int n = nBase + wn * 64 + nt * 16 + col0;
                const size_t idx = (size_t)m * N + n;
                const float v = acc[mi][nt][j];
                if constexpr (EPI == 0) {
                    ((unsigned short*)outv)[idx] = f2bf(v);
                } else if constexpr (EPI == 1) {
                    ((float*)outv)[idx] = v + resid[idx];
                } else {
                    const float sv = v * (1.0f / (1.0f + __expf(-v)));
                    ((unsigned short*)outv)[idx] = f2bf(sv * bf2f(gate[idx]));
                }
            }
        }
    }
}

// ---------------------------------------------------------------------------
// Flash attention: grid (S/128, H, B), 256 threads = 8 waves, wave = 16 q rows.
// q,k,v,o are bf16 [B*S, H*DH].  Causal, scale = 1/sqrt(DH).
// ---------------------------------------------------------------------------
__global__ __launch_bounds__(256) void attn_kernel(const unsigned short* __restrict__ qb,
                                                   const unsigned short* __restrict__ kb,
                                                   const unsigned short* __restrict__ vb,
                                                   unsigned short* __restrict__ ob) {
    __shared__ unsigned short kT[128 * 136];          // [key][dh]
    __shared__ unsigned short vT[128 * 136];          // [dh][key]  (transposed)
    __shared__ unsigned short pT[8 * 16 * 136];       // per-wave P slab [16][key]

    const int tid = threadIdx.x, lane = tid & 31, wave = tid >> 5;
    const int qt = blockIdx.x, h = blockIdx.y, b = blockIdx.z;
    const int qBase = qt * 128;
    const int rowStride = HH * DHH;                   // 2048
    const size_t baseBS = (size_t)b * SS * rowStride;
    const float scale = 0.088388347648318447f;        // 1/sqrt(128)

    // Q fragments held in registers for the whole kernel: 4 K-steps over DH=128.
    v16bf qfrag[4];
    {
        const unsigned short* qp =
            qb + baseBS + (size_t)(qBase + wave * 16) * rowStride + h * DHH;
#pragma unroll
        for (int i = 0; i < 4; i++) qfrag[i] = load_frag_a(qp + i * 32, rowStride, lane);
    }

    const v8f zero = {};
    v8f o_acc[8];
    float m_run[8], l_run[8];
#pragma unroll
    for (int j = 0; j < 8; j++) { m_run[j] = -1e30f; l_run[j] = 0.f; }
#pragma unroll
    for (int nt = 0; nt < 8; nt++) o_acc[nt] = zero;

    const int lhalf = (lane < 16) ? 0 : 8;
    const int col0  = lane & 15;
    const int nChunks = qt + 1;                       // causal

    for (int kc = 0; kc < nChunks; ++kc) {
        const int kBase = kc * 128;
        const unsigned short* kg = kb + baseBS + (size_t)kBase * rowStride + h * DHH;
        const unsigned short* vg = vb + baseBS + (size_t)kBase * rowStride + h * DHH;

        __syncthreads();   // previous iteration done with kT/vT

        // K tile: async global->LDS, row-major copy (8 x b128 per thread)
#pragma unroll
        for (int i = 0; i < 8; i++) {
            const int u  = tid + i * 256;
            const int rr = u >> 4;
            const int cc = (u & 15) * 8;
            async_copy16(kg + (size_t)rr * rowStride + cc, kT + rr * 136 + cc);
        }
        // V tile: transpose through VGPRs, two rows packed per dword store
#pragma unroll
        for (int i = 0; i < 4; i++) {
            const int u  = tid + i * 256;             // 0..1023
            const int rp = u >> 4;                    // row pair 0..63
            const int cc = (u & 15) * 8;
            const int r0 = rp * 2;
            U4 a = *(const U4*)(vg + (size_t)r0 * rowStride + cc);
            U4 bq = *(const U4*)(vg + (size_t)(r0 + 1) * rowStride + cc);
            const unsigned short* pa = (const unsigned short*)&a;
            const unsigned short* pb = (const unsigned short*)&bq;
#pragma unroll
            for (int e = 0; e < 8; e++) {
                const unsigned int d = (unsigned)pa[e] | ((unsigned)pb[e] << 16);
                *(unsigned int*)(vT + (cc + e) * 136 + r0) = d;
            }
        }
        wait_async<0>();
        __syncthreads();

        // scores S = Q K^T : 32 wmmas
        v8f s[8];
#pragma unroll
        for (int nt = 0; nt < 8; nt++) s[nt] = zero;
        {
            v16bf bcur = load_frag_b(kT, 136, lane);
#pragma unroll
            for (int it = 0; it < 32; ++it) {
                const int d0 = it >> 3, nt = it & 7;
                v16bf bnxt = bcur;
                if (it < 31) {
                    const int it1 = it + 1;
                    bnxt = load_frag_b(kT + (it1 & 7) * 16 * 136 + (it1 >> 3) * 32, 136,
                                       lane);
                }
                s[nt] = wmma_bf16(qfrag[d0], bcur, s[nt]);
                bcur = bnxt;
            }
        }

        // scale + causal mask + row max
        float nm[8];
#pragma unroll
        for (int j = 0; j < 8; j++) nm[j] = m_run[j];
#pragma unroll
        for (int nt = 0; nt < 8; nt++) {
            const int kIdx = kBase + nt * 16 + col0;
#pragma unroll
            for (int j = 0; j < 8; j++) {
                float v = s[nt][j] * scale;
                const int qIdx = qBase + wave * 16 + lhalf + j;
                if (kIdx > qIdx) v = -1e9f;
                s[nt][j] = v;
                nm[j] = fmaxf(nm[j], v);
            }
        }
#pragma unroll
        for (int j = 0; j < 8; j++) {
#pragma unroll
            for (int off = 1; off < 16; off <<= 1)
                nm[j] = fmaxf(nm[j], __shfl_xor(nm[j], off, 32));
        }

        // exp, row sum, spill P (bf16) to this wave's LDS slab
        float rs[8];
#pragma unroll
        for (int j = 0; j < 8; j++) rs[j] = 0.f;
        unsigned short* pw = pT + wave * 16 * 136;
#pragma unroll
        for (int nt = 0; nt < 8; nt++) {
#pragma unroll
            for (int j = 0; j < 8; j++) {
                const float p = __expf(s[nt][j] - nm[j]);
                rs[j] += p;
                pw[(lhalf + j) * 136 + nt * 16 + col0] = f2bf(p);
            }
        }
#pragma unroll
        for (int j = 0; j < 8; j++) {
#pragma unroll
            for (int off = 1; off < 16; off <<= 1) rs[j] += __shfl_xor(rs[j], off, 32);
            const float alpha = __expf(m_run[j] - nm[j]);
            l_run[j] = l_run[j] * alpha + rs[j];
            m_run[j] = nm[j];
#pragma unroll
            for (int nt = 0; nt < 8; nt++) o_acc[nt][j] *= alpha;
        }

        // wave-local LDS RAW: stores above feed the A-fragment loads below
        asm volatile("s_wait_dscnt 0x0" ::: "memory");

        // O += P V : K dim = key chunk (128), N dim = dh (128)
#pragma unroll
        for (int d0 = 0; d0 < 4; ++d0) {
            v16bf pa = load_frag_a(pw + d0 * 32, 136, lane);
            v16bf bcur = load_frag_b(vT + d0 * 32, 136, lane);
#pragma unroll
            for (int nt = 0; nt < 8; ++nt) {
                v16bf bnxt = bcur;
                if (nt < 7)
                    bnxt = load_frag_b(vT + (nt + 1) * 16 * 136 + d0 * 32, 136, lane);
                o_acc[nt] = wmma_bf16(pa, bcur, o_acc[nt]);
                bcur = bnxt;
            }
        }
    }

    // normalize and store
    unsigned short* op =
        ob + baseBS + (size_t)(qBase + wave * 16) * rowStride + h * DHH;
#pragma unroll
    for (int j = 0; j < 8; j++) {
        const float inv = 1.0f / l_run[j];
#pragma unroll
        for (int nt = 0; nt < 8; nt++)
            op[(size_t)(lhalf + j) * rowStride + nt * 16 + col0] =
                f2bf(o_acc[nt][j] * inv);
    }
}

// ---------------------------------------------------------------------------
// Host-side orchestration
// ---------------------------------------------------------------------------
extern "C" void kernel_launch(void* const* d_in, const int* in_sizes, int n_in,
                              void* d_out, int out_size, void* d_ws, size_t ws_size,
                              hipStream_t stream) {
    (void)in_sizes; (void)n_in; (void)out_size; (void)ws_size;
    const float* x  = (const float*)d_in[0];
    // d_in[1] = mask (causal; handled analytically)
    const float* wq = (const float*)d_in[2];
    const float* wk = (const float*)d_in[3];
    const float* wv = (const float*)d_in[4];
    const float* wo = (const float*)d_in[5];
    const float* w1 = (const float*)d_in[6];
    const float* w2 = (const float*)d_in[7];
    const float* w3 = (const float*)d_in[8];
    const float* gA = (const float*)d_in[9];
    const float* gF = (const float*)d_in[10];

    const size_t M = MM, D = DIMD, F = FFF;

    char* ws = (char*)d_ws;
    size_t off = 0;
    auto take = [&](size_t bytes) -> char* {
        char* p = ws + off;
        off = (off + bytes + 255) & ~(size_t)255;
        return p;
    };
    unsigned short* hbf  = (unsigned short*)take(M * D * 2);
    unsigned short* wqb  = (unsigned short*)take(D * D * 2);
    unsigned short* wkb  = (unsigned short*)take(D * D * 2);
    unsigned short* wvb  = (unsigned short*)take(D * D * 2);
    unsigned short* wob  = (unsigned short*)take(D * D * 2);
    unsigned short* w1b  = (unsigned short*)take(F * D * 2);
    unsigned short* w2b  = (unsigned short*)take(D * F * 2);
    unsigned short* w3b  = (unsigned short*)take(F * D * 2);
    unsigned short* qbf  = (unsigned short*)take(M * D * 2);
    unsigned short* kbf  = (unsigned short*)take(M * D * 2);
    unsigned short* vbf  = (unsigned short*)take(M * D * 2);
    unsigned short* obf  = (unsigned short*)take(M * D * 2);
    float*          x2f  = (float*)take(M * D * 4);
    unsigned short* h2bf = (unsigned short*)take(M * D * 2);
    unsigned short* gbf  = (unsigned short*)take(M * F * 2);   // gate, then ff (aliased)

    auto cast = [&](const float* src, unsigned short* dst, size_t n) {
        cast_bf16_kernel<<<dim3((unsigned)(n / 1024)), 256, 0, stream>>>(src, dst, (int)n);
    };
    cast(wq, wqb, D * D);
    cast(wk, wkb, D * D);
    cast(wv, wvb, D * D);
    cast(wo, wob, D * D);
    cast(w1, w1b, F * D);
    cast(w2, w2b, D * F);
    cast(w3, w3b, F * D);

    // h = rmsnorm(x, g_attn)
    rmsnorm_kernel<<<dim3((unsigned)M), 256, 0, stream>>>(x, gA, hbf);

    // q,k,v projections
    const dim3 gProj((unsigned)(D / 128), (unsigned)(M / 256));
    gemm_bf16_kernel<0><<<gProj, 256, 0, stream>>>(hbf, wqb, nullptr, nullptr, qbf,
                                                   (int)M, (int)D, (int)D);
    gemm_bf16_kernel<0><<<gProj, 256, 0, stream>>>(hbf, wkb, nullptr, nullptr, kbf,
                                                   (int)M, (int)D, (int)D);
    gemm_bf16_kernel<0><<<gProj, 256, 0, stream>>>(hbf, wvb, nullptr, nullptr, vbf,
                                                   (int)M, (int)D, (int)D);

    // RoPE on q and k (in place)
    const int ropeTotal = (int)(M * HH * 64);
    rope_kernel<<<dim3((unsigned)(ropeTotal / 256)), 256, 0, stream>>>(qbf, ropeTotal);
    rope_kernel<<<dim3((unsigned)(ropeTotal / 256)), 256, 0, stream>>>(kbf, ropeTotal);

    // causal flash attention
    attn_kernel<<<dim3(SS / 128, HH, BB), 256, 0, stream>>>(qbf, kbf, vbf, obf);

    // x2 = x + o @ wo^T
    gemm_bf16_kernel<1><<<gProj, 256, 0, stream>>>(obf, wob, x, nullptr, x2f,
                                                   (int)M, (int)D, (int)D);

    // h2 = rmsnorm(x2, g_ffn)
    rmsnorm_kernel<<<dim3((unsigned)M), 256, 0, stream>>>(x2f, gF, h2bf);

    // gate = h2 @ w3^T ; ff = silu(h2 @ w1^T) * gate (written over gate buffer)
    const dim3 gFF((unsigned)(F / 128), (unsigned)(M / 256));
    gemm_bf16_kernel<0><<<gFF, 256, 0, stream>>>(h2bf, w3b, nullptr, nullptr, gbf,
                                                 (int)M, (int)F, (int)D);
    gemm_bf16_kernel<2><<<gFF, 256, 0, stream>>>(h2bf, w1b, nullptr, gbf, gbf,
                                                 (int)M, (int)F, (int)D);

    // out = x2 + ff @ w2^T
    gemm_bf16_kernel<1><<<gProj, 256, 0, stream>>>(gbf, w2b, x2f, nullptr, d_out,
                                                   (int)M, (int)D, (int)F);
}
